// Graph2dConvolution_16509854286195
// MI455X (gfx1250) — matrix operand order repaired and checked by reading, not compiled, not run on hardware
//
#include <hip/hip_runtime.h>
#include <hip/hip_bf16.h>

typedef __attribute__((ext_vector_type(16))) __bf16 v16bf;
typedef __attribute__((ext_vector_type(8)))  float  v8f;

#define HDIM 256
#define WDIM 256
#define HWSZ 65536
#define CIN  64
#define COUT 64
#define KCL  16

__device__ __forceinline__ unsigned short f2bf_bits(float f) {
    unsigned int u = __builtin_bit_cast(unsigned int, f);
    unsigned int r = u + 0x7FFFu + ((u >> 16) & 1u);   // round-to-nearest-even
    return (unsigned short)(r >> 16);
}
__device__ __forceinline__ __bf16 f2bf(float f) {
    unsigned short h = f2bf_bits(f);
    return __builtin_bit_cast(__bf16, h);
}

// ---------------------------------------------------------------------------
// Kernel 0: zero the accumulator region of the workspace (sums + counts).
// Must run every launch: harness poisons ws once and never re-poisons.
// ---------------------------------------------------------------------------
__global__ void zero_ws_kernel(float* __restrict__ ws, int n) {
    int i = blockIdx.x * blockDim.x + threadIdx.x;
    if (i < n) ws[i] = 0.0f;
}

// ---------------------------------------------------------------------------
// Kernel 1: per-(batch,cluster) channel sums + counts via LDS atomics.
// Block = 1024 pixels of one batch; 512 blocks total. Memory-bound.
// ---------------------------------------------------------------------------
__global__ __launch_bounds__(256) void segsum_kernel(
    const float* __restrict__ x, const int* __restrict__ index,
    float* __restrict__ wsum, float* __restrict__ wcnt)
{
    __shared__ float lsum[KCL * CIN];
    __shared__ float lcnt[KCL];
    __shared__ int   iidx[1024];

    const int tid = threadIdx.x;
    const int b   = blockIdx.x >> 6;          // 64 blocks per batch
    const int pixbase = (blockIdx.x & 63) << 10;

    for (int e = tid; e < KCL * CIN; e += 256) lsum[e] = 0.0f;
    if (tid < KCL) lcnt[tid] = 0.0f;
    for (int e = tid; e < 1024; e += 256)
        iidx[e] = index[b * HWSZ + pixbase + e] & 15;
    __syncthreads();

    // channel-outer loop: 256 consecutive lanes hit 256 consecutive pixels
    for (int e = tid; e < CIN * 1024; e += 256) {
        int c = e >> 10;
        int i = e & 1023;
        float v = x[((b * CIN + c) << 16) + pixbase + i];
        atomicAdd(&lsum[iidx[i] * CIN + c], v);
    }
    for (int i = tid; i < 1024; i += 256)
        atomicAdd(&lcnt[iidx[i]], 1.0f);
    __syncthreads();

    for (int e = tid; e < KCL * CIN; e += 256) {
        int k = e >> 6, c = e & 63;
        atomicAdd(&wsum[(b * KCL + k) * CIN + c], lsum[k * CIN + c]);
    }
    if (tid < KCL) atomicAdd(&wcnt[b * KCL + tid], lcnt[tid]);
}

// ---------------------------------------------------------------------------
// Kernel 2: means, M = Wg*Wg^T, G = means*M*means^T,
// q[i,j] = G[ii]+G[jj]-2G[ij] (M symmetric), adj = exp(-q) off-diag,
// adj_means = adj @ means.  Tiny (~2 MFLOP): one block.
// ---------------------------------------------------------------------------
__global__ __launch_bounds__(256) void graph_stats_kernel(
    const float* __restrict__ Wg,
    const float* __restrict__ wsum, const float* __restrict__ wcnt,
    float* __restrict__ wadjm)
{
    __shared__ float WgS[CIN * CIN];
    __shared__ float Msh[CIN * CIN];
    __shared__ float mns[KCL * CIN];
    __shared__ float Psh[KCL * CIN];
    __shared__ float Gsh[KCL * KCL];
    __shared__ float Ash[KCL * KCL];

    const int tid = threadIdx.x;

    for (int e = tid; e < CIN * CIN; e += 256) WgS[e] = Wg[e];
    __syncthreads();

    // M[r][c] = dot(Wg[r,:], Wg[c,:])
    for (int e = tid; e < CIN * CIN; e += 256) {
        int r = e >> 6, c = e & 63;
        float s = 0.0f;
        for (int d = 0; d < CIN; ++d) s += WgS[r * CIN + d] * WgS[c * CIN + d];
        Msh[e] = s;
    }
    __syncthreads();

    for (int b = 0; b < 8; ++b) {
        for (int e = tid; e < KCL * CIN; e += 256) {
            int k = e >> 6, c = e & 63;
            float cnt = wcnt[b * KCL + k];
            mns[e] = wsum[(b * KCL + k) * CIN + c] / (cnt == 0.0f ? 1.0f : cnt);
        }
        __syncthreads();
        // P = means @ M   (16x64)
        for (int e = tid; e < KCL * CIN; e += 256) {
            int k = e >> 6, c = e & 63;
            float s = 0.0f;
            for (int d = 0; d < CIN; ++d) s += mns[k * CIN + d] * Msh[d * CIN + c];
            Psh[e] = s;
        }
        __syncthreads();
        // G = P @ means^T  (16x16)
        if (tid < KCL * KCL) {
            int i = tid >> 4, j = tid & 15;
            float s = 0.0f;
            for (int c = 0; c < CIN; ++c) s += Psh[i * CIN + c] * mns[j * CIN + c];
            Gsh[tid] = s;
        }
        __syncthreads();
        if (tid < KCL * KCL) {
            int i = tid >> 4, j = tid & 15;
            float q = Gsh[i * KCL + i] + Gsh[j * KCL + j] - 2.0f * Gsh[i * KCL + j];
            Ash[tid] = (i == j) ? 0.0f : __expf(-q);
        }
        __syncthreads();
        // adj_means = adj @ means  (16x64)
        for (int e = tid; e < KCL * CIN; e += 256) {
            int k = e >> 6, c = e & 63;
            float s = 0.0f;
            for (int j = 0; j < KCL; ++j) s += Ash[k * KCL + j] * mns[j * CIN + c];
            wadjm[b * KCL * CIN + e] = s;
        }
        __syncthreads();
    }
}

// ---------------------------------------------------------------------------
// Kernel 3: fused feat = x + adj_means[idx], then 3x3 conv as implicit GEMM
// with v_wmma_f32_16x16x32_bf16.
// Block tile: 16 rows x 16 cols x 64 Cout (2048 blocks, 256 threads).
// Each wave: M = 32 pixels (2 rows, two A-frags) x N = 64 -> B fragments
// are reused across both M-tiles: 8 WMMA per K-step from 16 A + 32 B dwords.
// GEMM: M = 256 pixels/block, N = 64, K = 576 (9 taps * 64 ch).
// ---------------------------------------------------------------------------
__global__ __launch_bounds__(256) void conv_wmma_kernel(
    const float* __restrict__ x, const int* __restrict__ index,
    const float* __restrict__ conv_w, const float* __restrict__ wadjm,
    float* __restrict__ out)
{
    // B-matrix: K-pairs packed per dword, row stride 68 dwords (bank-clean)
    __shared__ unsigned int Wlds[288 * 68];          // 78336 B
    // A-tile: halo 18x18 pixels x 64 ch, stride 66 bf16 (33 dwords, odd)
    __shared__ __bf16 Alds[324 * 66];                // 42768 B
    __shared__ float  adjm_l[KCL * CIN];             //  4096 B
    __shared__ int    iidx_l[324];                   //  1296 B

    const int tid = threadIdx.x;
    const int bx  = blockIdx.x;
    const int b   = bx >> 8;                 // 256 tiles per batch
    const int rem = bx & 255;
    const int h0  = (rem >> 4) << 4;         // tile row * 16
    const int w0  = (rem & 15) << 4;         // tile col * 16

    // warm GL2 for the input tile while we stage weights/indices
    {
        int rr = tid >> 4, cc = (tid & 15) << 2;     // 16 rows x 16 lanes
        int h = h0 + rr, w = w0 + cc;
        __builtin_prefetch(&x[((b * CIN) << 16) + (h << 8) + w], 0, 1);
    }

    for (int e = tid; e < KCL * CIN; e += 256)
        adjm_l[e] = wadjm[b * KCL * CIN + e];

    // stage weights: K = tap*64 + c (tap = kh*3+kw); dword = {bf16(K), bf16(K+1)}
    for (int e = tid; e < 288 * 64; e += 256) {
        int row = e >> 6;                    // K/2
        int n   = e & 63;
        int K   = row << 1;
        int tap = K >> 6;
        int c   = K & 63;
        float f0 = conv_w[(n * CIN + c) * 9 + tap];
        float f1 = conv_w[(n * CIN + c + 1) * 9 + tap];
        Wlds[row * 68 + n] =
            (unsigned int)f2bf_bits(f0) | ((unsigned int)f2bf_bits(f1) << 16);
    }

    // halo cluster indices (18x18)
    for (int e = tid; e < 324; e += 256) {
        int rr = e / 18, cc = e % 18;
        int h = h0 - 1 + rr, w = w0 - 1 + cc;
        int v = 0;
        if (h >= 0 && h < HDIM && w >= 0 && w < WDIM)
            v = index[b * HWSZ + (h << 8) + w] & 15;
        iidx_l[e] = v;
    }
    __syncthreads();

    // A-tile: feat = x + adj_means[idx], bf16, zero-padded borders
    for (int e = tid; e < 324 * 64; e += 256) {
        int pix = e % 324;                   // pixel-fast -> coalesced on w
        int c   = e / 324;
        int rr  = pix / 18, cc = pix % 18;
        int h = h0 - 1 + rr, w = w0 - 1 + cc;
        float f = 0.0f;
        if (h >= 0 && h < HDIM && w >= 0 && w < WDIM)
            f = x[((b * CIN + c) << 16) + (h << 8) + w]
                + adjm_l[iidx_l[pix] * CIN + c];
        Alds[pix * 66 + c] = f2bf(f);
    }
    __syncthreads();

    const int lane = tid & 31;
    const int wv   = tid >> 5;               // wave id: output rows 2wv, 2wv+1
    const int half = lane >> 4;
    const int mcol = lane & 15;              // A: M index = output col; B: N low bits

    v8f acc[8] = {};                         // [ridx*4 + t]

    for (int kk = 0; kk < 576; kk += 32) {
        // Two A fragments (rows 2wv and 2wv+1): 16-bit A 16x32 layout
        union { v16bf v; unsigned int u[8]; } af[2];
#pragma unroll
        for (int ridx = 0; ridx < 2; ++ridx) {
            int r = (wv << 1) + ridx;
#pragma unroll
            for (int v = 0; v < 8; ++v) {
                int kb  = 2 * v + (v < 4 ? 0 : 8) + 8 * half;
                int K0  = kk + kb;           // even
                int tap = K0 >> 6;
                int c   = K0 & 63;
                int kh  = tap / 3;
                int kw  = tap - kh * 3;
                int pix = (r + kh) * 18 + (mcol + kw);
                af[ridx].u[v] = *(const unsigned int*)&Alds[pix * 66 + c];
            }
        }
        int kg = half << 4;
#pragma unroll
        for (int t = 0; t < 4; ++t) {
            // B fragment: 16-bit B 32x16 layout; reused for both M-tiles
            union { v16bf v; unsigned int u[8]; } bfv;
            int n = (t << 4) + mcol;
#pragma unroll
            for (int v = 0; v < 8; ++v) {
                int K = kk + 2 * v + (v < 4 ? 0 : 8) + 8 * half;
                bfv.u[v] = Wlds[(K >> 1) * 68 + n];
            }
            acc[t] = __builtin_amdgcn_wmma_f32_16x16x32_bf16(
                false, af[0].v, false, bfv.v, (short)0, acc[t], false, false);
            acc[4 + t] = __builtin_amdgcn_wmma_f32_16x16x32_bf16(
                false, af[1].v, false, bfv.v, (short)0, acc[4 + t], false, false);
        }
    }

    // transpose result through LDS (alias weight region) for coalesced stores
    __syncthreads();
    float* Clds = (float*)Wlds;              // needs 256*65*4 = 66560 B
#pragma unroll
    for (int t = 0; t < 4; ++t) {
        int n = (t << 4) + mcol;
#pragma unroll
        for (int ridx = 0; ridx < 2; ++ridx) {
            int row = (wv << 1) + ridx;
#pragma unroll
            for (int g = 0; g < 8; ++g) {
                int m = g + (half << 3);     // C/D layout: VGPR g, half-split M
                int p = (row << 4) + m;
                Clds[p * 65 + n] = acc[ridx * 4 + t][g];
            }
        }
    }
    __syncthreads();
    for (int e = tid; e < 16384; e += 256) {
        int col = e & 15;
        int n   = (e >> 4) & 63;
        int row = e >> 10;
        out[((b * COUT + n) << 16) + ((h0 + row) << 8) + (w0 + col)]
            = Clds[(row * 16 + col) * 65 + n];
    }
}

// ---------------------------------------------------------------------------
extern "C" void kernel_launch(void* const* d_in, const int* in_sizes, int n_in,
                              void* d_out, int out_size, void* d_ws, size_t ws_size,
                              hipStream_t stream) {
    const float* x      = (const float*)d_in[0];   // (8,64,256,256)
    const int*   index  = (const int*)  d_in[1];   // (8,1,256,256)
    const float* Wg     = (const float*)d_in[2];   // (64,64)
    const float* conv_w = (const float*)d_in[3];   // (64,64,3,3)
    float*       out    = (float*)d_out;           // (8,64,256,256)

    float* wsum = (float*)d_ws;                    // 8*16*64 = 8192 floats
    float* wcnt = wsum + 8192;                     // 128 floats
    float* wadj = wsum + 8320;                     // 8192 floats

    zero_ws_kernel<<<(8320 + 255) / 256, 256, 0, stream>>>(wsum, 8320);
    segsum_kernel<<<512, 256, 0, stream>>>(x, index, wsum, wcnt);
    graph_stats_kernel<<<1, 256, 0, stream>>>(Wg, wsum, wcnt, wadj);
    conv_wmma_kernel<<<2048, 256, 0, stream>>>(x, index, conv_w, wadj, out);
}
